// MoAETrasnformerBlock_89850715833125
// MI455X (gfx1250) — matrix operand order repaired and checked by reading, not compiled, not run on hardware
//
#include <hip/hip_runtime.h>
#include <hip/hip_bf16.h>

// ---------------------------------------------------------------- constants
#define D_MODEL   1024
#define N_HEAD    16
#define N_HEAD_MOE 8
#define N_EXPERTS 8
#define BATCH     8
#define SEQ       1024
#define NTOK      (BATCH * SEQ)          // 8192 tokens
#define CAP       320                    // int(S*TOP_N*CAP_FACTOR/E)
#define EROWS     (BATCH * CAP)          // 2560 rows per expert
#define EBC       (N_EXPERTS * EROWS)    // 20480 expert slot rows

typedef __bf16 bf16_t;
typedef bf16_t bf16x16 __attribute__((ext_vector_type(16)));
typedef float  f32x8   __attribute__((ext_vector_type(8)));

__device__ __forceinline__ unsigned short f2bf(float f) {
    unsigned u = __float_as_uint(f);
    u += 0x7FFFu + ((u >> 16) & 1u);       // round-to-nearest-even
    return (unsigned short)(u >> 16);
}

union Frag { bf16x16 v; unsigned u[8]; };

// LDS byte offset of a generic pointer to __shared__ memory.
// Per CDNA5 ISA 10.2: flat addresses in the LDS aperture use addr[31:0] as the
// LDS address, so truncation of the generic address yields the DS offset.
__device__ __forceinline__ unsigned lds_u32(const void* p) {
    return (unsigned)(unsigned long long)p;
}

// CDNA5 async global->LDS copy (tracked by ASYNCcnt), see CDNA5 ISA
__device__ __forceinline__ void async_copy_b128(unsigned ldsoff, const void* g) {
    asm volatile("global_load_async_to_lds_b128 %0, %1, off"
                 :: "v"(ldsoff), "v"(g) : "memory");
}
__device__ __forceinline__ void wait_async0() {
    asm volatile("s_wait_asynccnt 0" ::: "memory");
}
__device__ __forceinline__ void wait_ds0() {
    asm volatile("s_wait_dscnt 0" ::: "memory");
}

// ---------------------------------------------------------------- utilities
__global__ void __launch_bounds__(256)
f32_to_bf16_kernel(const float* __restrict__ s, unsigned short* __restrict__ d, long long n) {
    long long stride = (long long)gridDim.x * blockDim.x;
    for (long long i = (long long)blockIdx.x * blockDim.x + threadIdx.x; i < n; i += stride)
        d[i] = f2bf(s[i]);
}

__global__ void __launch_bounds__(256)
zero_u32_kernel(unsigned* __restrict__ p, long long n) {
    long long stride = (long long)gridDim.x * blockDim.x;
    for (long long i = (long long)blockIdx.x * blockDim.x + threadIdx.x; i < n; i += stride)
        p[i] = 0u;
}

// ---------------------------------------------------------------- WMMA GEMM
// C[M,N] = A[M,K] (bf16, row-major) @ W[K,N] (bf16, row-major) + bias
// one 16x16 output tile per wave; batched over gridDim.y with given strides.
// A tile is staged LDS-side through the async copy units.
__global__ void __launch_bounds__(256)
wmma_gemm_kernel(const unsigned short* __restrict__ A, const unsigned short* __restrict__ W,
                 const float* __restrict__ bias, float* __restrict__ Cf,
                 unsigned short* __restrict__ Cb,
                 int M, int N, int K, long long sA, long long sW, long long sC) {
    __shared__ __attribute__((aligned(16))) unsigned short atile[8][16 * 32];
    int wave  = (int)((blockIdx.x * blockDim.x + threadIdx.x) >> 5);
    int lane  = (int)(threadIdx.x & 31);
    int wslot = (int)(threadIdx.x >> 5);
    int tilesN = N >> 4;
    int nTiles = (M >> 4) * tilesN;
    if (wave >= nTiles) return;                       // wave-uniform guard
    int tm = wave / tilesN, tn = wave % tilesN;
    long long aOff = (long long)blockIdx.y * sA;
    long long wOff = (long long)blockIdx.y * sW;
    long long cOff = (long long)blockIdx.y * sC;
    int ln = lane & 15, half = lane >> 4;

    const unsigned short* Atile = A + aOff + (long long)(tm * 16) * K;     // 16x32 tile rows
    const unsigned short* Wl    = W + wOff + tn * 16 + ln;                 // B: lane = N col
    unsigned abase = lds_u32(&atile[wslot][0]);
    int row0 = lane >> 2,        s0 = lane & 3;       // 64 x 16B segments, 2 per lane
    int row1 = (32 + lane) >> 2, s1 = (32 + lane) & 3;

    f32x8 acc = {};
    for (int k0 = 0; k0 < K; k0 += 32) {
        wait_ds0();                                   // WAR: previous frag reads done
        async_copy_b128(abase + (unsigned)(row0 * 32 + s0 * 8) * 2,
                        Atile + (long long)row0 * K + k0 + s0 * 8);
        async_copy_b128(abase + (unsigned)(row1 * 32 + s1 * 8) * 2,
                        Atile + (long long)row1 * K + k0 + s1 * 8);
        __builtin_prefetch((const void*)(Wl + (long long)(k0 + 32) * N), 0, 1);
        Frag af, bfm;
#pragma unroll
        for (int v = 0; v < 8; ++v) {
            int kb = ((v & 4) << 2) + half * 8 + (v & 3) * 2;              // B K-pair offset
            bfm.u[v] = (unsigned)Wl[(long long)(k0 + kb) * N] |
                       ((unsigned)Wl[(long long)(k0 + kb + 1) * N] << 16);
        }
        wait_async0();                                // RAW: A tile resident in LDS
#pragma unroll
        for (int v = 0; v < 8; ++v) {
            int kb = ((v & 4) << 2) + half * 8 + (v & 3) * 2;              // A K-pair offset
            af.u[v] = *(const unsigned*)&atile[wslot][ln * 32 + kb];
        }
        acc = __builtin_amdgcn_wmma_f32_16x16x32_bf16(false, af.v, false, bfm.v,
                                                      (short)0, acc, false, false);
    }
    int n = tn * 16 + ln;
    float bv = bias ? bias[n] : 0.0f;
    float*          cfp = Cf ? (Cf + cOff) : nullptr;
    unsigned short* cbp = Cb ? (Cb + cOff) : nullptr;
#pragma unroll
    for (int r = 0; r < 8; ++r) {
        int row = tm * 16 + r + half * 8;
        float vv = acc[r] + bv;
        long long off = (long long)row * N + n;
        if (cfp) cfp[off] = vv;
        if (cbp) cbp[off] = f2bf(vv);
    }
}

// ---------------------------------------------------------------- flash attention
// Layout: [BI, S, H*HD] bf16, one wave per (batch*head, 16-row query block).
// 32-key K tile async-staged into LDS; P re-layout bounced through LDS.
template <int HD>
__global__ void __launch_bounds__(256)
flash_attn_kernel(const unsigned short* __restrict__ Qb, const unsigned short* __restrict__ Kb,
                  const unsigned short* __restrict__ Vb, unsigned short* __restrict__ Ob,
                  int numBH, int H, int S, float scale) {
    constexpr int KC   = HD / 32;   // K-chunks for Q@K^T
    constexpr int OT   = HD / 16;   // N-tiles for P@V
    constexpr int SEGS = HD / 8;    // 16B segments per 32-key tile row pair-count
    __shared__ __attribute__((aligned(16))) unsigned short ktile[8][32 * HD];
    __shared__ __attribute__((aligned(16))) unsigned short ptile[8][16 * 32];

    int wave  = (int)((blockIdx.x * blockDim.x + threadIdx.x) >> 5);
    int lane  = (int)(threadIdx.x & 31);
    int wslot = (int)(threadIdx.x >> 5);
    int qBlocks = S >> 4;
    if (wave >= numBH * qBlocks) return;
    int bh = wave / qBlocks, qb = wave % qBlocks;
    int bi = bh / H, h = bh % H;
    const int Dm = H * HD;
    long long base = (long long)bi * S * Dm + (long long)h * HD;
    int ln = lane & 15, half = lane >> 4;
    unsigned kbase_lds = lds_u32(&ktile[wslot][0]);
    const unsigned short* Ktile0 = Kb + base;

    Frag qf[KC];
#pragma unroll
    for (int kc = 0; kc < KC; ++kc)
#pragma unroll
        for (int v = 0; v < 8; ++v) {
            int kd = kc * 32 + ((v & 4) << 2) + half * 8 + (v & 3) * 2;
            qf[kc].u[v] = *(const unsigned*)(Qb + base + (long long)(qb * 16 + ln) * Dm + kd);
        }

    f32x8 o[OT];
    f32x8 zf = {};
#pragma unroll
    for (int t = 0; t < OT; ++t) o[t] = zf;
    float m_run[8], l_run[8];
#pragma unroll
    for (int r = 0; r < 8; ++r) { m_run[r] = -1e30f; l_run[r] = 0.0f; }

    for (int kb = 0; kb < S; kb += 32) {
        wait_ds0();                                   // WAR on ktile/ptile
#pragma unroll
        for (int j = 0; j < SEGS; ++j) {              // stage 32 x HD bf16 K tile
            int id  = j * 32 + lane;
            int row = id / SEGS;                      // key within block (0..31)
            int seg = id % SEGS;                      // 16B segment within row
            async_copy_b128(kbase_lds + (unsigned)(row * HD + seg * 8) * 2,
                            Ktile0 + (long long)(kb + row) * Dm + seg * 8);
        }
        wait_async0();
        f32x8 s0 = {}, s1 = {};
#pragma unroll
        for (int kc = 0; kc < KC; ++kc) {
            Frag b0, b1;
#pragma unroll
            for (int v = 0; v < 8; ++v) {             // B lane = N (key), elems = K (head-dim)
                int kd = kc * 32 + ((v & 4) << 2) + half * 8 + (v & 3) * 2;
                b0.u[v] = *(const unsigned*)&ktile[wslot][ln * HD + kd];
                b1.u[v] = *(const unsigned*)&ktile[wslot][(16 + ln) * HD + kd];
            }
            s0 = __builtin_amdgcn_wmma_f32_16x16x32_bf16(false, qf[kc].v, false, b0.v,
                                                         (short)0, s0, false, false);
            s1 = __builtin_amdgcn_wmma_f32_16x16x32_bf16(false, qf[kc].v, false, b1.v,
                                                         (short)0, s1, false, false);
        }
        // online softmax (row stats replicated across each 16-lane half)
        float p0[8], p1[8];
#pragma unroll
        for (int r = 0; r < 8; ++r) {
            float a = s0[r] * scale, b = s1[r] * scale;
            float rm = fmaxf(a, b);
            rm = fmaxf(rm, __shfl_xor(rm, 1, 32));
            rm = fmaxf(rm, __shfl_xor(rm, 2, 32));
            rm = fmaxf(rm, __shfl_xor(rm, 4, 32));
            rm = fmaxf(rm, __shfl_xor(rm, 8, 32));
            float mnew = fmaxf(m_run[r], rm);
            float e0 = __expf(a - mnew), e1 = __expf(b - mnew);
            float rs = e0 + e1;
            rs += __shfl_xor(rs, 1, 32);
            rs += __shfl_xor(rs, 2, 32);
            rs += __shfl_xor(rs, 4, 32);
            rs += __shfl_xor(rs, 8, 32);
            float alpha = __expf(m_run[r] - mnew);
            l_run[r] = l_run[r] * alpha + rs;
            m_run[r] = mnew;
            p0[r] = e0; p1[r] = e1;
#pragma unroll
            for (int t = 0; t < OT; ++t) o[t][r] *= alpha;
        }
        // C-layout P -> LDS -> A-layout P (intra-wave, LDS is in-order per wave)
#pragma unroll
        for (int r = 0; r < 8; ++r) {
            int row = r + 8 * half;
            ptile[wslot][row * 32 + ln]      = f2bf(p0[r]);
            ptile[wslot][row * 32 + 16 + ln] = f2bf(p1[r]);
        }
        wait_ds0();
        Frag pf;
#pragma unroll
        for (int v = 0; v < 8; ++v) {
            int kk = ((v & 4) << 2) + half * 8 + (v & 3) * 2;
            pf.u[v] = *(const unsigned*)&ptile[wslot][ln * 32 + kk];
        }
#pragma unroll
        for (int t = 0; t < OT; ++t) {
            Frag vf;
#pragma unroll
            for (int v = 0; v < 8; ++v) {  // B lane = N (head-dim col), K-pairs (keys) in table order
                int kk = ((v & 4) << 2) + half * 8 + (v & 3) * 2;
                vf.u[v] = (unsigned)Vb[base + (long long)(kb + kk) * Dm + t * 16 + ln] |
                          ((unsigned)Vb[base + (long long)(kb + kk + 1) * Dm + t * 16 + ln] << 16);
            }
            o[t] = __builtin_amdgcn_wmma_f32_16x16x32_bf16(false, pf.v, false, vf.v,
                                                           (short)0, o[t], false, false);
        }
    }
    float invl[8];
#pragma unroll
    for (int r = 0; r < 8; ++r) invl[r] = 1.0f / l_run[r];
#pragma unroll
    for (int t = 0; t < OT; ++t)
#pragma unroll
        for (int r = 0; r < 8; ++r) {
            int row = qb * 16 + r + 8 * half;
            Ob[base + (long long)row * Dm + t * 16 + ln] = f2bf(o[t][r] * invl[r]);
        }
}

// ---------------------------------------------------------------- add + layernorm
__global__ void __launch_bounds__(256)
add_ln_kernel(const float* __restrict__ x, const float* __restrict__ y,
              const float* __restrict__ g, const float* __restrict__ bta,
              float* __restrict__ of, unsigned short* __restrict__ ob) {
    __shared__ float rs[8], rs2[8];
    int t = (int)blockIdx.x;
    const float* xr = x + (long long)t * D_MODEL;
    const float* yr = y + (long long)t * D_MODEL;
    float vals[4], s = 0.f, s2 = 0.f;
#pragma unroll
    for (int j = 0; j < 4; ++j) {
        int i = (int)threadIdx.x + j * 256;
        float v = xr[i] + yr[i];
        vals[j] = v; s += v; s2 += v * v;
    }
#pragma unroll
    for (int m = 16; m >= 1; m >>= 1) { s += __shfl_xor(s, m, 32); s2 += __shfl_xor(s2, m, 32); }
    int lane = (int)(threadIdx.x & 31), w = (int)(threadIdx.x >> 5);
    if (lane == 0) { rs[w] = s; rs2[w] = s2; }
    __syncthreads();
    float ts = 0.f, ts2 = 0.f;
#pragma unroll
    for (int k = 0; k < 8; ++k) { ts += rs[k]; ts2 += rs2[k]; }
    float mu  = ts * (1.0f / D_MODEL);
    float var = ts2 * (1.0f / D_MODEL) - mu * mu;
    float inv = rsqrtf(var + 1e-5f);
#pragma unroll
    for (int j = 0; j < 4; ++j) {
        int i = (int)threadIdx.x + j * 256;
        float vo = (vals[j] - mu) * inv * g[i] + bta[i];
        of[(long long)t * D_MODEL + i] = vo;
        if (ob) ob[(long long)t * D_MODEL + i] = f2bf(vo);
    }
}

// ---------------------------------------------------------------- router (wave per token)
__global__ void __launch_bounds__(256)
router_kernel(const float* __restrict__ x1, const float* __restrict__ gateW,
              int* __restrict__ idx0, int* __restrict__ idx1,
              float* __restrict__ g0, float* __restrict__ g1, int* __restrict__ thr1,
              float* __restrict__ probs_buf, float* __restrict__ lse2_buf) {
    int wave = (int)((blockIdx.x * blockDim.x + threadIdx.x) >> 5);
    int lane = (int)(threadIdx.x & 31);
    if (wave >= NTOK) return;
    const float* xr = x1 + (long long)wave * D_MODEL;
    float lg[N_EXPERTS];
#pragma unroll
    for (int e = 0; e < N_EXPERTS; ++e) lg[e] = 0.f;
    for (int i = lane; i < D_MODEL; i += 32) {
        float xv = xr[i];
#pragma unroll
        for (int e = 0; e < N_EXPERTS; ++e) lg[e] += xv * gateW[i * N_EXPERTS + e];
    }
#pragma unroll
    for (int e = 0; e < N_EXPERTS; ++e) {
        float v = lg[e];
        v += __shfl_xor(v, 1, 32); v += __shfl_xor(v, 2, 32);
        v += __shfl_xor(v, 4, 32); v += __shfl_xor(v, 8, 32); v += __shfl_xor(v, 16, 32);
        lg[e] = v;
    }
    float mx = lg[0];
#pragma unroll
    for (int e = 1; e < N_EXPERTS; ++e) mx = fmaxf(mx, lg[e]);
    float pr[N_EXPERTS], se = 0.f;
#pragma unroll
    for (int e = 0; e < N_EXPERTS; ++e) { pr[e] = __expf(lg[e] - mx); se += pr[e]; }
    float inv = 1.0f / se;
#pragma unroll
    for (int e = 0; e < N_EXPERTS; ++e) pr[e] *= inv;
    int i0 = 0;
#pragma unroll
    for (int e = 1; e < N_EXPERTS; ++e) if (pr[e] > pr[i0]) i0 = e;
    int i1 = (i0 == 0) ? 1 : 0;
#pragma unroll
    for (int e = 0; e < N_EXPERTS; ++e) if (e != i1 && e != i0 && pr[e] > pr[i1]) i1 = e;
    if (lane == 0) {
        idx0[wave] = i0; idx1[wave] = i1;
        g0[wave] = pr[i0]; g1[wave] = pr[i1];
        thr1[wave] = (pr[i1] > 0.2f) ? 1 : 0;
#pragma unroll
        for (int e = 0; e < N_EXPERTS; ++e) probs_buf[(long long)wave * N_EXPERTS + e] = pr[e];
        float lse = mx + __logf(se);
        lse2_buf[wave] = lse * lse;
    }
}

// ---------------------------------------------------------------- capacity scan
__global__ void scan_kernel(const int* __restrict__ idx0, const int* __restrict__ idx1,
                            const int* __restrict__ thr1,
                            int* __restrict__ pos0, int* __restrict__ pos1,
                            int* __restrict__ keep0, int* __restrict__ keep1) {
    int t = (int)threadIdx.x;
    if (t >= BATCH * N_EXPERTS) return;
    int b = t >> 3, e = t & 7;
    int base = b * SEQ;
    int c0 = 0;
    for (int s = 0; s < SEQ; ++s) {
        int i = base + s;
        if (idx0[i] == e) {
            pos0[i] = (c0 < CAP) ? c0 : (CAP - 1);
            keep0[i] = (c0 < CAP) ? 1 : 0;
            c0++;
        }
    }
    int c1 = 0;
    for (int s = 0; s < SEQ; ++s) {
        int i = base + s;
        if (idx1[i] == e) {
            int p = c0 + c1;
            int kt = thr1[i];
            pos1[i] = (p < CAP) ? p : (CAP - 1);
            keep1[i] = (kt && p < CAP) ? 1 : 0;
            if (kt) c1++;
        }
    }
}

// ---------------------------------------------------------------- dispatch
__global__ void __launch_bounds__(256)
dispatch_kernel(const float* __restrict__ x1, const int* __restrict__ idx,
                const int* __restrict__ pos, const int* __restrict__ keep,
                unsigned short* __restrict__ ein) {
    int t = (int)blockIdx.x;
    if (!keep[t]) return;
    int b = t / SEQ;
    long long dst = ((((long long)idx[t] * BATCH + b) * CAP) + pos[t]) * D_MODEL;
    const float* src = x1 + (long long)t * D_MODEL;
    for (int i = (int)threadIdx.x; i < D_MODEL; i += 256)
        ein[dst + i] = f2bf(src[i]);
}

// ---------------------------------------------------------------- combine + LN2
__global__ void __launch_bounds__(256)
combine_ln_kernel(const float* __restrict__ x1, const float* __restrict__ eout,
                  const int* __restrict__ idx0, const int* __restrict__ pos0,
                  const int* __restrict__ keep0, const float* __restrict__ g0,
                  const int* __restrict__ idx1, const int* __restrict__ pos1,
                  const int* __restrict__ keep1, const float* __restrict__ g1,
                  const float* __restrict__ lg, const float* __restrict__ lb,
                  float* __restrict__ out) {
    __shared__ float rs[8], rs2[8];
    int t = (int)blockIdx.x;
    int b = t / SEQ;
    float w0 = keep0[t] ? g0[t] : 0.f;
    float w1 = keep1[t] ? g1[t] : 0.f;
    long long o0 = ((((long long)idx0[t] * BATCH + b) * CAP) + pos0[t]) * D_MODEL;
    long long o1 = ((((long long)idx1[t] * BATCH + b) * CAP) + pos1[t]) * D_MODEL;
    const float* xr = x1 + (long long)t * D_MODEL;
    float vals[4], s = 0.f, s2 = 0.f;
#pragma unroll
    for (int j = 0; j < 4; ++j) {
        int i = (int)threadIdx.x + j * 256;
        float v = xr[i] + w0 * eout[o0 + i] + w1 * eout[o1 + i];
        vals[j] = v; s += v; s2 += v * v;
    }
#pragma unroll
    for (int m = 16; m >= 1; m >>= 1) { s += __shfl_xor(s, m, 32); s2 += __shfl_xor(s2, m, 32); }
    int lane = (int)(threadIdx.x & 31), w = (int)(threadIdx.x >> 5);
    if (lane == 0) { rs[w] = s; rs2[w] = s2; }
    __syncthreads();
    float ts = 0.f, ts2 = 0.f;
#pragma unroll
    for (int k = 0; k < 8; ++k) { ts += rs[k]; ts2 += rs2[k]; }
    float mu  = ts * (1.0f / D_MODEL);
    float var = ts2 * (1.0f / D_MODEL) - mu * mu;
    float inv = rsqrtf(var + 1e-5f);
#pragma unroll
    for (int j = 0; j < 4; ++j) {
        int i = (int)threadIdx.x + j * 256;
        out[(long long)t * D_MODEL + i] = (vals[j] - mu) * inv * lg[i] + lb[i];
    }
}

// ---------------------------------------------------------------- deterministic loss reduce
__global__ void __launch_bounds__(256)
loss_kernel(const float* __restrict__ probs_buf, const float* __restrict__ lse2_buf,
            const int* __restrict__ idx0, float* __restrict__ tail) {
    __shared__ float sbuf[17][256];
    float loc[17];
#pragma unroll
    for (int k = 0; k < 17; ++k) loc[k] = 0.f;
    for (int t = (int)threadIdx.x; t < NTOK; t += 256) {
        loc[idx0[t]] += 1.0f;                               // f counts [0..7]
#pragma unroll
        for (int e = 0; e < N_EXPERTS; ++e)                 // P sums   [8..15]
            loc[8 + e] += probs_buf[(long long)t * N_EXPERTS + e];
        loc[16] += lse2_buf[t];                             // z sum
    }
#pragma unroll
    for (int k = 0; k < 17; ++k) sbuf[k][threadIdx.x] = loc[k];
    __syncthreads();
    if (threadIdx.x < 17) {
        float sum = 0.f;
        for (int j = 0; j < 256; ++j) sum += sbuf[threadIdx.x][j];
        sbuf[threadIdx.x][0] = sum;
    }
    __syncthreads();
    if (threadIdx.x == 0) {
        const float invN = 1.0f / (float)NTOK;
        float bal = 0.f;
        for (int e = 0; e < N_EXPERTS; ++e)
            bal += (sbuf[e][0] * invN) * (sbuf[8 + e][0] * invN);
        bal *= (float)N_EXPERTS;
        float z = sbuf[16][0] * invN;
        float balL = 0.01f * bal, zL = 0.001f * z;
        tail[0] = balL + zL; tail[1] = balL; tail[2] = zL;
    }
}

// ---------------------------------------------------------------- host
extern "C" void kernel_launch(void* const* d_in, const int* in_sizes, int n_in,
                              void* d_out, int out_size, void* d_ws, size_t ws_size,
                              hipStream_t stream) {
    (void)in_sizes; (void)n_in; (void)out_size; (void)ws_size;
    const float* x     = (const float*)d_in[0];
    const float* Wq    = (const float*)d_in[1];
    const float* Wk    = (const float*)d_in[2];
    const float* Wv    = (const float*)d_in[3];
    const float* Wo    = (const float*)d_in[4];
    const float* bq    = (const float*)d_in[5];
    const float* bk    = (const float*)d_in[6];
    const float* bv    = (const float*)d_in[7];
    const float* bo    = (const float*)d_in[8];
    const float* ln1_g = (const float*)d_in[9];
    const float* ln1_b = (const float*)d_in[10];
    const float* ln2_g = (const float*)d_in[11];
    const float* ln2_b = (const float*)d_in[12];
    const float* gateW = (const float*)d_in[13];
    const float* eWq   = (const float*)d_in[14];
    const float* eWk   = (const float*)d_in[15];
    const float* eWv   = (const float*)d_in[16];
    const float* eWo   = (const float*)d_in[17];
    float* out = (float*)d_out;

    char* wp = (char*)d_ws;
    auto carve = [&](size_t bytes) { void* r = (void*)wp; wp += (bytes + 255) & ~(size_t)255; return r; };
    const long long ND  = (long long)NTOK * D_MODEL;           // 8 Mi elems
    const long long ED  = (long long)EBC  * D_MODEL;           // 20 Mi elems
    const long long DD  = (long long)D_MODEL * D_MODEL;
    const long long EDD = (long long)N_EXPERTS * DD;

    unsigned short* xb   = (unsigned short*)carve(ND * 2);
    unsigned short* wqb  = (unsigned short*)carve(DD * 2);
    unsigned short* wkb  = (unsigned short*)carve(DD * 2);
    unsigned short* wvb  = (unsigned short*)carve(DD * 2);
    unsigned short* wob  = (unsigned short*)carve(DD * 2);
    unsigned short* ewqb = (unsigned short*)carve(EDD * 2);
    unsigned short* ewkb = (unsigned short*)carve(EDD * 2);
    unsigned short* ewvb = (unsigned short*)carve(EDD * 2);
    unsigned short* ewob = (unsigned short*)carve(EDD * 2);
    unsigned short* Qb   = (unsigned short*)carve(ED * 2);
    unsigned short* Kb   = (unsigned short*)carve(ED * 2);
    unsigned short* Vb   = (unsigned short*)carve(ED * 2);
    unsigned short* Ob   = (unsigned short*)carve(ED * 2);
    float*          attn = (float*)carve(ND * 4);
    float*          x1   = (float*)carve(ND * 4);
    unsigned short* x1b  = (unsigned short*)carve(ND * 2);
    unsigned short* ein  = (unsigned short*)carve(ED * 2);
    float*          eout = (float*)carve(ED * 4);
    int*   idx0 = (int*)carve(NTOK * 4);
    int*   idx1 = (int*)carve(NTOK * 4);
    int*   pos0 = (int*)carve(NTOK * 4);
    int*   pos1 = (int*)carve(NTOK * 4);
    int*   kp0  = (int*)carve(NTOK * 4);
    int*   kp1  = (int*)carve(NTOK * 4);
    int*   thr1 = (int*)carve(NTOK * 4);
    float* gg0  = (float*)carve(NTOK * 4);
    float* gg1  = (float*)carve(NTOK * 4);
    float* pbuf = (float*)carve((long long)NTOK * N_EXPERTS * 4);
    float* zbuf = (float*)carve(NTOK * 4);

    // --- bf16 conversions
    f32_to_bf16_kernel<<<2048, 256, 0, stream>>>(x, xb, ND);
    f32_to_bf16_kernel<<<1024, 256, 0, stream>>>(Wq, wqb, DD);
    f32_to_bf16_kernel<<<1024, 256, 0, stream>>>(Wk, wkb, DD);
    f32_to_bf16_kernel<<<1024, 256, 0, stream>>>(Wv, wvb, DD);
    f32_to_bf16_kernel<<<1024, 256, 0, stream>>>(Wo, wob, DD);
    f32_to_bf16_kernel<<<2048, 256, 0, stream>>>(eWq, ewqb, EDD);
    f32_to_bf16_kernel<<<2048, 256, 0, stream>>>(eWk, ewkb, EDD);
    f32_to_bf16_kernel<<<2048, 256, 0, stream>>>(eWv, ewvb, EDD);
    f32_to_bf16_kernel<<<2048, 256, 0, stream>>>(eWo, ewob, EDD);
    zero_u32_kernel<<<2048, 256, 0, stream>>>((unsigned*)ein, ED / 2);

    // --- MHA: QKV projections (8192x1024x1024 each), flash attention, output proj
    wmma_gemm_kernel<<<4096, 256, 0, stream>>>(xb, wqb, bq, nullptr, Qb,
                                               NTOK, D_MODEL, D_MODEL, 0, 0, 0);
    wmma_gemm_kernel<<<4096, 256, 0, stream>>>(xb, wkb, bk, nullptr, Kb,
                                               NTOK, D_MODEL, D_MODEL, 0, 0, 0);
    wmma_gemm_kernel<<<4096, 256, 0, stream>>>(xb, wvb, bv, nullptr, Vb,
                                               NTOK, D_MODEL, D_MODEL, 0, 0, 0);
    flash_attn_kernel<64><<<1024, 256, 0, stream>>>(Qb, Kb, Vb, Ob,
                                                    BATCH * N_HEAD, N_HEAD, SEQ, 0.125f);
    wmma_gemm_kernel<<<4096, 256, 0, stream>>>(Ob, wob, bo, attn, nullptr,
                                               NTOK, D_MODEL, D_MODEL, 0, 0, 0);
    add_ln_kernel<<<NTOK, 256, 0, stream>>>(x, attn, ln1_g, ln1_b, x1, x1b);

    // --- routing
    router_kernel<<<1024, 256, 0, stream>>>(x1, gateW, idx0, idx1, gg0, gg1, thr1, pbuf, zbuf);
    scan_kernel<<<1, 64, 0, stream>>>(idx0, idx1, thr1, pos0, pos1, kp0, kp1);
    dispatch_kernel<<<NTOK, 256, 0, stream>>>(x1, idx0, pos0, kp0, ein);
    dispatch_kernel<<<NTOK, 256, 0, stream>>>(x1, idx1, pos1, kp1, ein);

    // --- experts: batched projections + flash attention (hd=128) + output proj
    const long long sA = (long long)EROWS * D_MODEL;
    dim3 ge(1280, N_EXPERTS, 1);
    wmma_gemm_kernel<<<ge, 256, 0, stream>>>(ein, ewqb, nullptr, nullptr, Qb,
                                             EROWS, D_MODEL, D_MODEL, sA, DD, sA);
    wmma_gemm_kernel<<<ge, 256, 0, stream>>>(ein, ewkb, nullptr, nullptr, Kb,
                                             EROWS, D_MODEL, D_MODEL, sA, DD, sA);
    wmma_gemm_kernel<<<ge, 256, 0, stream>>>(ein, ewvb, nullptr, nullptr, Vb,
                                             EROWS, D_MODEL, D_MODEL, sA, DD, sA);
    flash_attn_kernel<128><<<1280, 256, 0, stream>>>(Qb, Kb, Vb, Ob,
                                                     N_EXPERTS * BATCH * N_HEAD_MOE,
                                                     N_HEAD_MOE, CAP, 0.08838834764831845f);
    wmma_gemm_kernel<<<ge, 256, 0, stream>>>(Ob, ewob, nullptr, eout, nullptr,
                                             EROWS, D_MODEL, D_MODEL, sA, DD, sA);

    // --- combine + LN2 + losses
    combine_ln_kernel<<<NTOK, 256, 0, stream>>>(x1, eout, idx0, pos0, kp0, gg0,
                                                idx1, pos1, kp1, gg1, ln2_g, ln2_b, out);
    loss_kernel<<<1, 256, 0, stream>>>(pbuf, zbuf, idx0, out + ND);
}
